// RnnCoreModel_1142461301082
// MI455X (gfx1250) — matrix operand order repaired and checked by reading, not compiled, not run on hardware
//
#include <hip/hip_runtime.h>
#include <hip/hip_bf16.h>
#include <stdint.h>

#define S_LEN 4096
#define BATCH 8
#define HID 512
#define G4 2048  // 4*HID

typedef float v8f  __attribute__((ext_vector_type(8)));
typedef float v4f  __attribute__((ext_vector_type(4)));
typedef __bf16 v16bf __attribute__((ext_vector_type(16)));
typedef __bf16 v4bf  __attribute__((ext_vector_type(4)));
typedef unsigned int v4u __attribute__((ext_vector_type(4)));
typedef int v8i __attribute__((ext_vector_type(8)));
typedef int v4i __attribute__((ext_vector_type(4)));

#if __has_builtin(__builtin_amdgcn_tensor_load_to_lds) && __has_builtin(__builtin_amdgcn_s_wait_tensorcnt)
#define HAVE_TDM 1
#else
#define HAVE_TDM 0
#endif
#if __has_builtin(__builtin_amdgcn_cluster_id_x)
#define HAVE_CID 1
#else
#define HAVE_CID 0
#endif

// Native hardware float->bf16 (RNE) conversion.
static __device__ __forceinline__ unsigned short f2bf(float f) {
  __bf16 h = (__bf16)f;
  return __builtin_bit_cast(unsigned short, h);
}
static __device__ __forceinline__ float bf2f(unsigned short h) {
  return __uint_as_float((unsigned)h << 16);
}
static __device__ __forceinline__ float sigmoidf_(float x) {
  return 1.0f / (1.0f + __expf(-x));
}

union FragB { v4u u[2]; v16bf b; unsigned short h[16]; };

// 8 bf16 at p0 -> elems 0..7, 8 bf16 at p1 -> elems 8..15
static __device__ __forceinline__ v16bf load_frag_bf16(const unsigned short* p0,
                                                       const unsigned short* p1) {
  FragB f;
  f.u[0] = *(const v4u*)p0;
  f.u[1] = *(const v4u*)p1;
  return f.b;
}

#define WMMA_BF16(a, b, c) \
  __builtin_amdgcn_wmma_f32_16x16x32_bf16(false, (a), false, (b), (short)0, (c), false, false)

#if HAVE_TDM
// 2D TDM load: tile_h rows x tile_w elems (2-byte), global row stride = row_stride elems.
static __device__ __forceinline__ void tdm_load_2d(unsigned lds_addr, const void* gptr,
                                                   unsigned tile_w, unsigned tile_h,
                                                   unsigned row_stride) {
  unsigned long long ga = (unsigned long long)(uintptr_t)gptr;
  v4u g0;
  g0[0] = 1u;                                                   // count=1 valid descriptor
  g0[1] = lds_addr;                                             // bits 63:32 lds_addr
  g0[2] = (unsigned)ga;                                         // global_addr lo
  g0[3] = (unsigned)((ga >> 32) & 0x01FFFFFFu) | 0x80000000u;   // addr hi | type=2
  // group1 (256b) assembled as 4x u64
  unsigned long long q0 = (1ull << 16)                               // data_size = 2 bytes
                        | ((unsigned long long)(tile_w & 0xFFFFu) << 48);  // tensor_dim0 lo16
  unsigned long long q1 = ((unsigned long long)tile_h << 16)         // tensor_dim1 (bits 111:80)
                        | ((unsigned long long)(tile_w & 0xFFFFu) << 48);  // tile_dim0
  unsigned long long q2 = (unsigned long long)(tile_h & 0xFFFFu)     // tile_dim1
                        | ((unsigned long long)row_stride << 32);    // tensor_dim0_stride lo32
  unsigned long long q3 = 0ull;                                      // stride hi + dim1_stride
  v8i g1;
  g1[0] = (int)q0; g1[1] = (int)(q0 >> 32);
  g1[2] = (int)q1; g1[3] = (int)(q1 >> 32);
  g1[4] = (int)q2; g1[5] = (int)(q2 >> 32);
  g1[6] = (int)q3; g1[7] = (int)(q3 >> 32);
  v4i gz  = {0, 0, 0, 0};
  v8i gz8 = {0, 0, 0, 0, 0, 0, 0, 0};
  // 6-arg toolchain variant: (g0, g1, g2, g3, g4, cpol)
  __builtin_amdgcn_tensor_load_to_lds(g0, g1, gz, gz, gz8, 0);
}
#endif

// Cluster barrier (16 WGs) with monotonic-counter fallback for non-cluster launch.
static __device__ __forceinline__ void xbarrier(unsigned int* bar, unsigned target, int cid) {
  __syncthreads();
  if (cid != 0) {
#if HAVE_CID
    __builtin_amdgcn_s_cluster_barrier();
#endif
  } else {
    if (threadIdx.x == 0) {
      __hip_atomic_fetch_add(bar, 1u, __ATOMIC_RELEASE, __HIP_MEMORY_SCOPE_AGENT);
      while (__hip_atomic_load(bar, __ATOMIC_ACQUIRE, __HIP_MEMORY_SCOPE_AGENT) < target)
        __builtin_amdgcn_s_sleep(1);
    }
    __syncthreads();
  }
}

// ---------------------------------------------------------------------------
// Kernel 0: convert world-states (32768 x 512 f32) to bf16, vectorized x4.
// ---------------------------------------------------------------------------
__global__ void cvt_ws_kernel(const float* __restrict__ A, unsigned short* __restrict__ Ab) {
  size_t i = ((size_t)blockIdx.x * blockDim.x + threadIdx.x) * 4;
  v4f a = *(const v4f*)(A + i);
  v4bf h;
#pragma unroll
  for (int k = 0; k < 4; ++k) h[k] = (__bf16)a[k];
  *(v4bf*)(Ab + i) = h;
}

// ---------------------------------------------------------------------------
// Kernel 1: weight conversion / small precompute / state init.
// ---------------------------------------------------------------------------
__global__ void prep_kernel(
    const float* __restrict__ emb, const float* __restrict__ Wp,
    const float* __restrict__ Wih0, const float* __restrict__ Whh0,
    const float* __restrict__ bih0, const float* __restrict__ bhh0,
    const float* __restrict__ Wih1, const float* __restrict__ Whh1,
    const float* __restrict__ bih1, const float* __restrict__ bhh1,
    unsigned short* __restrict__ Wp_b, unsigned short* __restrict__ Wih0p_b,
    unsigned short* __restrict__ Whh0_b, unsigned short* __restrict__ Wih1_b,
    unsigned short* __restrict__ Whh1_b, float* __restrict__ gtab,
    float* __restrict__ b1sum, unsigned short* __restrict__ hx,
    unsigned int* __restrict__ bar) {
  long long i = (long long)blockIdx.x * blockDim.x + threadIdx.x;
  const long long NP0 = 960LL * 512, NP1 = 2048LL * 960, NP2 = 2048LL * 512,
                  NP3 = NP2, NP4 = NP2, NP5 = 32LL * 2048, NP6 = 2048, NP7 = 8192, NP8 = 64;
  if (i < NP0) { Wp_b[i] = f2bf(Wp[i]); return; } i -= NP0;
  if (i < NP1) { long long j = i / 960, k = i % 960; Wih0p_b[i] = f2bf(Wih0[j * 1024 + k]); return; } i -= NP1;
  if (i < NP2) { Whh0_b[i] = f2bf(Whh0[i]); return; } i -= NP2;
  if (i < NP3) { Wih1_b[i] = f2bf(Wih1[i]); return; } i -= NP3;
  if (i < NP4) { Whh1_b[i] = f2bf(Whh1[i]); return; } i -= NP4;
  if (i < NP5) {                      // gtab[a][j] = b0[j] + emb[a] . Wih0[j, 960:1024]
    long long a = i >> 11, j = i & 2047;
    float s = bih0[j] + bhh0[j];
#pragma unroll 8
    for (int k = 0; k < 64; ++k) s += emb[a * 64 + k] * Wih0[j * 1024 + 960 + k];
    gtab[i] = s; return;
  } i -= NP5;
  if (i < NP6) { b1sum[i] = bih1[i] + bhh1[i]; return; } i -= NP6;
  if (i < NP7) { hx[i] = 0; return; } i -= NP7;   // zero h0x + h1x (contiguous)
  if (i < NP8) { bar[i] = 0; return; }
}

// ---------------------------------------------------------------------------
// Kernel 2: proj = bf16(relu(ws_bf @ Wp^T + bp)).  M=32768, K=512, N=960.
// Block = 8 waves; wave -> 16 rows x 64 cols.  Pure bf16 fragment loads.
// ---------------------------------------------------------------------------
__global__ __launch_bounds__(256, 2) void proj_gemm_kernel(
    const unsigned short* __restrict__ A, const unsigned short* __restrict__ Wb,
    const float* __restrict__ bias, unsigned short* __restrict__ C) {
  const int wave = threadIdx.x >> 5, lane = threadIdx.x & 31;
  const int m0 = blockIdx.x * 128 + wave * 16;
  const int n0 = blockIdx.y * 64;
  const int r16 = lane & 15;
  const int kb_a = (lane >> 4) * 16, kb_b = (lane >> 4) * 16;
  const unsigned short* Arow = A + (size_t)(m0 + r16) * 512;

  v8f acc[4];
#pragma unroll
  for (int nt = 0; nt < 4; ++nt)
#pragma unroll
    for (int r = 0; r < 8; ++r) acc[nt][r] = 0.0f;

  for (int kt = 0; kt < 16; ++kt) {
    const int k0 = kt * 32;
    v16bf a = load_frag_bf16(Arow + k0 + kb_a, Arow + k0 + kb_a + 8);
#pragma unroll
    for (int nt = 0; nt < 4; ++nt) {
      const unsigned short* Brow = Wb + (size_t)(n0 + nt * 16 + r16) * 512 + k0 + kb_b;
      acc[nt] = WMMA_BF16(a, load_frag_bf16(Brow, Brow + 8), acc[nt]);
    }
  }
#pragma unroll
  for (int nt = 0; nt < 4; ++nt)
#pragma unroll
    for (int r = 0; r < 8; ++r) {
      int m = m0 + r + 8 * (lane >> 4);
      int n = n0 + nt * 16 + r16;
      float v = acc[nt][r] + bias[n];
      v = v > 0.0f ? v : 0.0f;
      C[(size_t)m * 960 + n] = f2bf(v);
    }
}

// ---------------------------------------------------------------------------
// Kernel 3: G0[t][b][:] = bf16(proj @ Wih0p^T + gtab[action]).  M=32768,K=960,N=2048.
// Output stored t-major for sequential streaming.
// ---------------------------------------------------------------------------
__global__ __launch_bounds__(256, 2) void g0_gemm_kernel(
    const unsigned short* __restrict__ A, const unsigned short* __restrict__ Wb,
    const float* __restrict__ gtab, const int* __restrict__ aidx,
    unsigned short* __restrict__ G0) {
  const int wave = threadIdx.x >> 5, lane = threadIdx.x & 31;
  const int m0 = blockIdx.x * 128 + wave * 16;
  const int n0 = blockIdx.y * 64;
  const int r16 = lane & 15;
  const int kb_a = (lane >> 4) * 16, kb_b = (lane >> 4) * 16;
  const unsigned short* Arow = A + (size_t)(m0 + r16) * 960;

  v8f acc[4];
#pragma unroll
  for (int nt = 0; nt < 4; ++nt)
#pragma unroll
    for (int r = 0; r < 8; ++r) acc[nt][r] = 0.0f;

  for (int kt = 0; kt < 30; ++kt) {
    const int k0 = kt * 32;
    v16bf a = load_frag_bf16(Arow + k0 + kb_a, Arow + k0 + kb_a + 8);
#pragma unroll
    for (int nt = 0; nt < 4; ++nt) {
      const unsigned short* Brow = Wb + (size_t)(n0 + nt * 16 + r16) * 960 + k0 + kb_b;
      acc[nt] = WMMA_BF16(a, load_frag_bf16(Brow, Brow + 8), acc[nt]);
    }
  }
#pragma unroll
  for (int r = 0; r < 8; ++r) {
    int row = m0 + r + 8 * (lane >> 4);
    int b = row >> 12, t = row & 4095;
    int ai = aidx[row];
#pragma unroll
    for (int nt = 0; nt < 4; ++nt) {
      int n = n0 + nt * 16 + r16;
      float v = acc[nt][r] + gtab[(size_t)ai * G4 + n];
      G0[((size_t)t * BATCH + b) * G4 + n] = f2bf(v);
    }
  }
}

// ---------------------------------------------------------------------------
// Kernel 4: persistent 16-WG cluster LSTM scan.  WG w owns h-columns [32w,32w+32).
// Wave v (0..7): gate = v>>1, 16 gate columns.  Per step:
//   L0: gates = G0slice(TDM-prefetched) + h0 @ Whh0_sliceT -> h0,c0 slice
//   cluster barrier ; gather full h0 from L2
//   L1: gates = b1 + h0 @ Wih1_sliceT + h1 @ Whh1_sliceT -> h1,c1 slice -> rnn_out
//   cluster barrier ; gather full h1
// ---------------------------------------------------------------------------
__global__ __launch_bounds__(256, 1) void lstm_seq_kernel(
    const unsigned short* __restrict__ G0,
    const unsigned short* __restrict__ Whh0,
    const unsigned short* __restrict__ Wih1,
    const unsigned short* __restrict__ Whh1,
    const float* __restrict__ b1sum,
    unsigned short* __restrict__ h0x,
    unsigned short* __restrict__ h1x,
    unsigned int* __restrict__ bar,
    float* __restrict__ out) {
  __shared__ __align__(16) unsigned short h0f[16][HID];
  __shared__ __align__(16) unsigned short h1f[16][HID];
  __shared__ __align__(16) unsigned short gbuf[2][4][BATCH][32];
  __shared__ float gact[4][BATCH][32];
  __shared__ float c0s[BATCH][32];
  __shared__ float c1s[BATCH][32];

  const int w = blockIdx.x;
  const int tid = threadIdx.x;
  const int wave = tid >> 5, lane = tid & 31;
  const int gate = wave >> 1;
  const int sub = (wave & 1) * 16;
  const int n16 = lane & 15;
  const bool lo = lane < 16;
  const int gcol0 = gate * HID + w * 32 + sub;
  const int kb_a = (lane >> 4) * 16, kb_b = (lane >> 4) * 16;
  const int cm = tid >> 5, ccol = tid & 31;    // cell-phase (batch,col) ownership
#if HAVE_CID
  const int cid = __builtin_amdgcn_cluster_id_x();
#else
  const int cid = 0;
#endif

  for (int i = tid; i < 16 * HID; i += 256) { h0f[i >> 9][i & 511] = 0; h1f[i >> 9][i & 511] = 0; }
  c0s[cm][ccol] = 0.0f;
  c1s[cm][ccol] = 0.0f;
  __syncthreads();

  const unsigned short* Wr0 = Whh0 + (size_t)(gcol0 + n16) * HID;
  const unsigned short* Wr1 = Wih1 + (size_t)(gcol0 + n16) * HID;
  const unsigned short* Wr2 = Whh1 + (size_t)(gcol0 + n16) * HID;

#if HAVE_TDM
  if (wave == 0) {                                // prime double buffer for t=0
    unsigned lds0 = (unsigned)(uintptr_t)&gbuf[0][0][0][0];
    const unsigned short* gsrc = G0 + (size_t)w * 32;
    for (int g = 0; g < 4; ++g) tdm_load_2d(lds0 + g * 512, gsrc + g * HID, 32, BATCH, G4);
  }
#endif

  unsigned bcnt = 0;
  for (int t = 0; t < S_LEN; ++t) {
    const int cur = t & 1;
#if HAVE_TDM
    if (wave == 0) __builtin_amdgcn_s_wait_tensorcnt(0);
    __syncthreads();
    if (wave == 0 && t + 1 < S_LEN) {             // prefetch next step's gate slice
      unsigned lds0 = (unsigned)(uintptr_t)&gbuf[cur ^ 1][0][0][0];
      const unsigned short* gsrc = G0 + (size_t)(t + 1) * BATCH * G4 + w * 32;
      for (int g = 0; g < 4; ++g) tdm_load_2d(lds0 + g * 512, gsrc + g * HID, 32, BATCH, G4);
    }
#else
    for (int i = tid; i < 4 * BATCH * 32; i += 256) {
      int g = i >> 8, r = (i >> 5) & 7, c = i & 31;
      gbuf[cur][g][r][c] = G0[(size_t)t * BATCH * G4 + (size_t)r * G4 + g * HID + w * 32 + c];
    }
    __syncthreads();
#endif

    // ---------------- layer 0 ----------------
    v8f acc;
#pragma unroll
    for (int r = 0; r < 8; ++r) acc[r] = lo ? bf2f(gbuf[cur][gate][r][sub + n16]) : 0.0f;
#pragma unroll
    for (int kt = 0; kt < 16; ++kt) {
      const int k0 = kt * 32;
      v16bf a = load_frag_bf16(&h0f[n16][k0 + kb_a], &h0f[n16][k0 + kb_a + 8]);
      v16bf b = load_frag_bf16(Wr0 + k0 + kb_b, Wr0 + k0 + kb_b + 8);
      acc = WMMA_BF16(a, b, acc);
    }
    if (lo) {
#pragma unroll
      for (int r = 0; r < 8; ++r)
        gact[gate][r][sub + n16] = (gate == 2) ? tanhf(acc[r]) : sigmoidf_(acc[r]);
    }
    __syncthreads();
    {
      float i_ = gact[0][cm][ccol], f_ = gact[1][cm][ccol];
      float g_ = gact[2][cm][ccol], o_ = gact[3][cm][ccol];
      float c = f_ * c0s[cm][ccol] + i_ * g_;
      c0s[cm][ccol] = c;
      h0x[cm * HID + w * 32 + ccol] = f2bf(o_ * tanhf(c));
    }
    __threadfence();
    xbarrier(bar, (++bcnt) * 16u, cid);
    {                                              // gather full h0 (rows 0..7)
      int base = tid * 16;
      *(v4u*)&h0f[base >> 9][base & 511]       = *(const v4u*)&h0x[base];
      *(v4u*)&h0f[base >> 9][(base & 511) + 8] = *(const v4u*)&h0x[base + 8];
    }
    __syncthreads();

    // ---------------- layer 1 ----------------
    const float bb = lo ? b1sum[gcol0 + n16] : 0.0f;
#pragma unroll
    for (int r = 0; r < 8; ++r) acc[r] = bb;
#pragma unroll
    for (int kt = 0; kt < 16; ++kt) {
      const int k0 = kt * 32;
      v16bf a = load_frag_bf16(&h0f[n16][k0 + kb_a], &h0f[n16][k0 + kb_a + 8]);
      v16bf b = load_frag_bf16(Wr1 + k0 + kb_b, Wr1 + k0 + kb_b + 8);
      acc = WMMA_BF16(a, b, acc);
    }
#pragma unroll
    for (int kt = 0; kt < 16; ++kt) {
      const int k0 = kt * 32;
      v16bf a = load_frag_bf16(&h1f[n16][k0 + kb_a], &h1f[n16][k0 + kb_a + 8]);
      v16bf b = load_frag_bf16(Wr2 + k0 + kb_b, Wr2 + k0 + kb_b + 8);
      acc = WMMA_BF16(a, b, acc);
    }
    if (lo) {
#pragma unroll
      for (int r = 0; r < 8; ++r)
        gact[gate][r][sub + n16] = (gate == 2) ? tanhf(acc[r]) : sigmoidf_(acc[r]);
    }
    __syncthreads();
    {
      float i_ = gact[0][cm][ccol], f_ = gact[1][cm][ccol];
      float g_ = gact[2][cm][ccol], o_ = gact[3][cm][ccol];
      float c = f_ * c1s[cm][ccol] + i_ * g_;
      c1s[cm][ccol] = c;
      float h = o_ * tanhf(c);
      h1x[cm * HID + w * 32 + ccol] = f2bf(h);
      out[(size_t)cm * S_LEN * HID + (size_t)t * HID + w * 32 + ccol] = h;
    }
    __threadfence();
    xbarrier(bar, (++bcnt) * 16u, cid);
    {
      int base = tid * 16;
      *(v4u*)&h1f[base >> 9][base & 511]       = *(const v4u*)&h1x[base];
      *(v4u*)&h1f[base >> 9][(base & 511) + 8] = *(const v4u*)&h1x[base + 8];
    }
    __syncthreads();
  }

  // final hidden/cell states: out = [rnn_out | hidden(2,8,512) | cell(2,8,512)]
  {
    const size_t base = (size_t)BATCH * S_LEN * HID;
    const int hc = w * 32 + ccol;
    out[base + (size_t)cm * HID + hc]                 = bf2f(h0f[cm][hc]);
    out[base + 4096 + (size_t)cm * HID + hc]          = bf2f(h1f[cm][hc]);
    out[base + 8192 + (size_t)cm * HID + hc]          = c0s[cm][ccol];
    out[base + 8192 + 4096 + (size_t)cm * HID + hc]   = c1s[cm][ccol];
  }
}

// ---------------------------------------------------------------------------
extern "C" void kernel_launch(void* const* d_in, const int* in_sizes, int n_in,
                              void* d_out, int out_size, void* d_ws, size_t ws_size,
                              hipStream_t stream) {
  (void)in_sizes; (void)n_in; (void)out_size; (void)ws_size;
  const float* ws_in = (const float*)d_in[0];
  const int*   aidx  = (const int*)d_in[1];
  const float* emb   = (const float*)d_in[2];
  const float* Wp    = (const float*)d_in[3];
  const float* bp    = (const float*)d_in[4];
  const float* Wih0  = (const float*)d_in[5];
  const float* Whh0  = (const float*)d_in[6];
  const float* bih0  = (const float*)d_in[7];
  const float* bhh0  = (const float*)d_in[8];
  const float* Wih1  = (const float*)d_in[9];
  const float* Whh1  = (const float*)d_in[10];
  const float* bih1  = (const float*)d_in[11];
  const float* bhh1  = (const float*)d_in[12];

  char* p = (char*)d_ws;
  auto take = [&](size_t bytes) -> void* {
    void* r = (void*)p;
    p += (bytes + 255) & ~(size_t)255;
    return r;
  };
  unsigned short* Wp_b    = (unsigned short*)take((size_t)960 * 512 * 2);
  unsigned short* Wih0p_b = (unsigned short*)take((size_t)2048 * 960 * 2);
  unsigned short* Whh0_b  = (unsigned short*)take((size_t)2048 * 512 * 2);
  unsigned short* Wih1_b  = (unsigned short*)take((size_t)2048 * 512 * 2);
  unsigned short* Whh1_b  = (unsigned short*)take((size_t)2048 * 512 * 2);
  float*          gtab    = (float*)take((size_t)32 * 2048 * 4);
  float*          b1s     = (float*)take((size_t)2048 * 4);
  unsigned short* h0x     = (unsigned short*)take((size_t)8 * 512 * 2);  // 8192B, stays contiguous
  unsigned short* h1x     = (unsigned short*)take((size_t)8 * 512 * 2);  // with h0x for zeroing
  unsigned int*   bar     = (unsigned int*)take((size_t)64 * 4);
  unsigned short* ws_b    = (unsigned short*)take((size_t)32768 * 512 * 2);
  unsigned short* proj_b  = (unsigned short*)take((size_t)32768 * 960 * 2);
  unsigned short* G0      = (unsigned short*)take((size_t)4096 * 8 * 2048 * 2);

  // 32768*512 / (256*4) = 16384 blocks
  cvt_ws_kernel<<<16384, 256, 0, stream>>>(ws_in, ws_b);
  // total prep items = 491520+1966080+3*1048576+65536+2048+8192+64 = 5679168
  prep_kernel<<<22185, 256, 0, stream>>>(emb, Wp, Wih0, Whh0, bih0, bhh0,
                                         Wih1, Whh1, bih1, bhh1,
                                         Wp_b, Wih0p_b, Whh0_b, Wih1_b, Whh1_b,
                                         gtab, b1s, h0x, bar);
  proj_gemm_kernel<<<dim3(256, 15), 256, 0, stream>>>(ws_b, Wp_b, bp, proj_b);
  g0_gemm_kernel<<<dim3(256, 32), 256, 0, stream>>>(proj_b, Wih0p_b, gtab, aidx, G0);
  lstm_seq_kernel<<<16, 256, 0, stream>>>(G0, Whh0_b, Wih1_b, Whh1_b, b1s,
                                          h0x, h1x, bar, (float*)d_out);
}
